// FastVCompressor_65077344469073
// MI455X (gfx1250) — matrix operand order repaired and checked by reading, not compiled, not run on hardware
//
#include <hip/hip_runtime.h>

#define H_DIM 2048
#define HH_DIM 1024
#define C_DIM 32
#define N_ROWS 16384
#define MT 64           // rows per workgroup in assign_kernel
#define KP 2064         // padded LDS row stride (bf16 elems) for keys tile
#define HS 72           // padded LDS row stride for hidden chunk (64 cols + 8)

typedef __attribute__((ext_vector_type(16))) __bf16 v16bf;
typedef __attribute__((ext_vector_type(8)))  __bf16 v8bf;
typedef __attribute__((ext_vector_type(8)))  float  v8f;

union ABu { v16bf v; v8bf h[2]; };

__device__ __forceinline__ v8f wmma_bf16(const ABu& a, const ABu& b, v8f c) {
  return __builtin_amdgcn_wmma_f32_16x16x32_bf16(false, a.v, false, b.v,
                                                 (short)0, c, false, false);
}

// ---------------------------------------------------------------- utilities
__global__ __launch_bounds__(256) void cvt_kernel(const float* __restrict__ s,
                                                  __bf16* __restrict__ d, int n) {
  int i = blockIdx.x * 256 + threadIdx.x;
  if (i < n) d[i] = (__bf16)s[i];
}

__global__ __launch_bounds__(256) void zero_kernel(float* __restrict__ p, int n) {
  int i = blockIdx.x * 256 + threadIdx.x;
  if (i < n) p[i] = 0.0f;
}

// ---------------------------------------------------------------- kernel 1
// Fused: hidden = ReLU(keys @ cw1.T + cb1); scores = hidden @ cw2.T + cb2;
// idx = argmax(scores).  One WG handles 64 rows; keys tile cached in LDS bf16.
// K-loop is software pipelined; each B tile is reused by two row-tiles.
// waves_per_eu(2): cap at 512 VGPRs/wave so the 8-wave WG always fits
// (2 waves per SIMD32, 2x512 = full 1024-VGPR file).
__global__ __launch_bounds__(256)
__attribute__((amdgpu_waves_per_eu(2)))
void assign_kernel(
    const float* __restrict__ keys, const __bf16* __restrict__ cw1b,
    const float* __restrict__ cb1, const __bf16* __restrict__ cw2b,
    const float* __restrict__ cb2, int* __restrict__ idx_out)
{
  extern __shared__ char smem[];
  __bf16* skeys  = (__bf16*)smem;                                    // [MT][KP]
  __bf16* shid   = (__bf16*)(smem + (size_t)MT * KP * 2);            // [MT][HS]
  float*  sscore = (float*)(smem + (size_t)MT * KP * 2 + MT * HS * 2); // [MT][32]

  const int tid  = threadIdx.x;
  const int wave = tid >> 5;
  const int lane = tid & 31;
  const int ln16 = lane & 15;
  const bool hi  = lane >= 16;
  const size_t m0g = (size_t)blockIdx.x * MT;

  // ---- stage keys tile to LDS, fp32 -> bf16, packed b64 stores ----
  for (int r = wave; r < MT; r += 8) {
    const float* src = keys + (m0g + r) * H_DIM;
    __bf16* dst = skeys + r * KP;
    for (int k = lane * 4; k < H_DIM; k += 128) {
      float4 f = *(const float4*)(src + k);
      union { __bf16 b[4]; unsigned long long u; } p;
      p.b[0] = (__bf16)f.x; p.b[1] = (__bf16)f.y;
      p.b[2] = (__bf16)f.z; p.b[3] = (__bf16)f.w;
      *(unsigned long long*)(dst + k) = p.u;
    }
  }
  __syncthreads();

  // hidden-stage roles: row-group (32 rows = 2 tiles) x col tile (16 of 64)
  const int mg  = wave & 1;
  const int nt  = wave >> 1;
  const int mt0 = mg * 32, mt1 = mg * 32 + 16;
  // scores-stage roles: row tile (16 of 64) x col half (16 of 32)
  const int smt = (wave >> 1) * 16;
  const int sch = (wave & 1) * 16;

  v8f sacc = {};
  const __bf16* aK0 = skeys + (size_t)(mt0 + ln16) * KP + (hi ? 8 : 0);
  const __bf16* aK1 = aK0 + (size_t)16 * KP;

  for (int hc = 0; hc < HH_DIM; hc += 64) {
    // ---- hidden[64, hc:hc+64]: 2 WMMAs (2 row tiles) per shared B tile ----
    {
      v8f acc0 = {}, acc1 = {};
      const int bn = hc + nt * 16 + ln16;                 // hidden unit = cw1 row
      const __bf16* bp = cw1b + (size_t)bn * H_DIM + (hi ? 8 : 0);
      ABu a0c, a1c, bc, a0n, a1n, bnx;
      a0c.h[0] = *(const v8bf*)(aK0);      a0c.h[1] = *(const v8bf*)(aK0 + 16);
      a1c.h[0] = *(const v8bf*)(aK1);      a1c.h[1] = *(const v8bf*)(aK1 + 16);
      bc.h[0]  = *(const v8bf*)(bp);       bc.h[1]  = *(const v8bf*)(bp + 16);
#pragma unroll 2
      for (int k = 0; k < H_DIM - 32; k += 32) {
        const int kn = k + 32;
        a0n.h[0] = *(const v8bf*)(aK0 + kn); a0n.h[1] = *(const v8bf*)(aK0 + kn + 16);
        a1n.h[0] = *(const v8bf*)(aK1 + kn); a1n.h[1] = *(const v8bf*)(aK1 + kn + 16);
        bnx.h[0] = *(const v8bf*)(bp + kn);  bnx.h[1] = *(const v8bf*)(bp + kn + 16);
        acc0 = wmma_bf16(a0c, bc, acc0);
        acc1 = wmma_bf16(a1c, bc, acc1);
        a0c = a0n; a1c = a1n; bc = bnx;
      }
      acc0 = wmma_bf16(a0c, bc, acc0);
      acc1 = wmma_bf16(a1c, bc, acc1);

      const int hcol = nt * 16 + ln16;
      const float bias = cb1[hc + hcol];
      const int rb0 = mt0 + (hi ? 8 : 0), rb1 = mt1 + (hi ? 8 : 0);
#pragma unroll
      for (int i = 0; i < 8; ++i) {
        float v0 = acc0[i] + bias, v1 = acc1[i] + bias;
        shid[(rb0 + i) * HS + hcol] = (__bf16)(v0 > 0.0f ? v0 : 0.0f);
        shid[(rb1 + i) * HS + hcol] = (__bf16)(v1 > 0.0f ? v1 : 0.0f);
      }
    }
    __syncthreads();
    // ---- scores[64,32] += hidden_chunk @ cw2_chunk.T (2 k-chunks) ----
#pragma unroll
    for (int kk = 0; kk < 64; kk += 32) {
      ABu a, b;
      const __bf16* ap = shid + (smt + ln16) * HS + kk + (hi ? 8 : 0);
      a.h[0] = *(const v8bf*)(ap);
      a.h[1] = *(const v8bf*)(ap + 16);
      const __bf16* bp2 = cw2b + (size_t)(sch + ln16) * HH_DIM + hc + kk + (hi ? 8 : 0);
      b.h[0] = *(const v8bf*)(bp2);
      b.h[1] = *(const v8bf*)(bp2 + 16);
      sacc = wmma_bf16(a, b, sacc);
    }
    __syncthreads();
  }

  // ---- bias + store scores, per-row argmax ----
  {
    const int scol = sch + ln16;
    const float b2 = cb2[scol];
    const int rb = smt + (hi ? 8 : 0);
#pragma unroll
    for (int i = 0; i < 8; ++i)
      sscore[(rb + i) * 32 + scol] = sacc[i] + b2;
  }
  __syncthreads();
  if (tid < MT) {
    const float* s = sscore + tid * 32;
    float best = s[0]; int bi = 0;
#pragma unroll
    for (int c = 1; c < C_DIM; ++c) { float v = s[c]; if (v > best) { best = v; bi = c; } }
    idx_out[m0g + tid] = bi;
  }
}

// ---------------------------------------------------------------- kernel 2
// Segment sums via race-free LDS accumulation (each (c,h) owned by one thread)
// then global f32 atomics.
__global__ __launch_bounds__(256) void segsum_kernel(
    const float* __restrict__ keys, const float* __restrict__ values,
    const int* __restrict__ idx, float* __restrict__ ksum,
    float* __restrict__ vsum, float* __restrict__ counts, int rows_per_wg)
{
  extern __shared__ float lds[];               // [2][C][512]
  float* lk = lds;
  float* lv = lds + C_DIM * 512;
  const int tid = threadIdx.x;
  const int r0 = blockIdx.x * rows_per_wg;

  for (int r = r0 + tid; r < r0 + rows_per_wg; r += 256)
    atomicAdd(&counts[idx[r]], 1.0f);

  for (int hc = 0; hc < H_DIM; hc += 512) {
    for (int i = tid; i < C_DIM * 512; i += 256) { lk[i] = 0.0f; lv[i] = 0.0f; }
    __syncthreads();
    for (int r = r0; r < r0 + rows_per_wg; ++r) {
      const int c = idx[r];
      const float* kp = keys   + (size_t)r * H_DIM + hc;
      const float* vp = values + (size_t)r * H_DIM + hc;
      for (int h = tid; h < 512; h += 256) {     // thread t owns h = t, t+256 only
        lk[c * 512 + h] += kp[h];
        lv[c * 512 + h] += vp[h];
      }
    }
    __syncthreads();
    for (int i = tid; i < C_DIM * 512; i += 256) {
      const int c = i >> 9, h = i & 511;
      atomicAdd(&ksum[(size_t)c * H_DIM + hc + h], lk[i]);
      atomicAdd(&vsum[(size_t)c * H_DIM + hc + h], lv[i]);
    }
    __syncthreads();
  }
}

// ---------------------------------------------------------------- kernel 3
__global__ __launch_bounds__(256) void finalize_kernel(
    const float* __restrict__ ksum, const float* __restrict__ vsum,
    const float* __restrict__ counts, const float* __restrict__ noise_k,
    const float* __restrict__ noise_v, __bf16* __restrict__ kcb,
    __bf16* __restrict__ vcb)
{
  const int i = blockIdx.x * 256 + threadIdx.x;       // over C*H
  const int h = i & (H_DIM - 1);
  float best = counts[0]; int src = 0;
#pragma unroll
  for (int j = 1; j < C_DIM; ++j) { float v = counts[j]; if (v > best) { best = v; src = j; } }
  const float cnt = counts[i >> 11];
  float km, vm;
  if (cnt == 0.0f) {
    const float d = fmaxf(best, 1.0f);
    km = ksum[(size_t)src * H_DIM + h] / d + 0.1f * noise_k[i];
    vm = vsum[(size_t)src * H_DIM + h] / d + 0.1f * noise_v[i];
  } else {
    km = ksum[i] / cnt;
    vm = vsum[i] / cnt;
  }
  kcb[i] = (__bf16)km;
  vcb[i] = (__bf16)vm;
}

// ---------------------------------------------------------------- WMMA tile
// 16x16 output tile, K-contraction, software pipelined (A,B from global/L2).
__device__ __forceinline__ v8f gemm32_tile(const __bf16* __restrict__ A,
                                           const __bf16* __restrict__ W,
                                           int K, int mt, int ncol0, int lane)
{
  const int ln16 = lane & 15;
  const bool hi = lane >= 16;
  const __bf16* ap = A + (size_t)(mt + ln16) * K + (hi ? 8 : 0);
  const __bf16* bp = W + (size_t)(ncol0 + ln16) * K + (hi ? 8 : 0);
  v8f acc = {};
  ABu ac, bc, an, bn;
  ac.h[0] = *(const v8bf*)(ap);     ac.h[1] = *(const v8bf*)(ap + 16);
  bc.h[0] = *(const v8bf*)(bp);     bc.h[1] = *(const v8bf*)(bp + 16);
#pragma unroll 2
  for (int k = 0; k < K - 32; k += 32) {
    const int kn = k + 32;
    an.h[0] = *(const v8bf*)(ap + kn); an.h[1] = *(const v8bf*)(ap + kn + 16);
    bn.h[0] = *(const v8bf*)(bp + kn); bn.h[1] = *(const v8bf*)(bp + kn + 16);
    acc = wmma_bf16(ac, bc, acc);
    ac = an; bc = bn;
  }
  acc = wmma_bf16(ac, bc, acc);
  return acc;
}

// ---------------------------------------------------------------- transform
__global__ __launch_bounds__(256)
__attribute__((amdgpu_waves_per_eu(4)))
void transform1_kernel(
    const __bf16* __restrict__ kc, const __bf16* __restrict__ vc,
    const __bf16* __restrict__ tw1b, const float* __restrict__ tb1,
    __bf16* __restrict__ hk, __bf16* __restrict__ hv)
{
  const int wave = threadIdx.x >> 5, lane = threadIdx.x & 31;
  const __bf16* A = blockIdx.y ? vc : kc;
  __bf16* O = blockIdx.y ? hv : hk;
  const int mt = (wave & 1) * 16;
  const int ncol0 = blockIdx.x * 64 + (wave >> 1) * 16;
  v8f acc = gemm32_tile(A, tw1b, H_DIM, mt, ncol0, lane);
  const int col = ncol0 + (lane & 15);
  const float bias = tb1[col];
  const int rb = mt + ((lane >= 16) ? 8 : 0);
#pragma unroll
  for (int i = 0; i < 8; ++i) {
    float v = acc[i] + bias;
    O[(size_t)(rb + i) * H_DIM + col] = (__bf16)(v > 0.0f ? v : 0.0f);
  }
}

__global__ __launch_bounds__(256)
__attribute__((amdgpu_waves_per_eu(4)))
void transform2_kernel(
    const __bf16* __restrict__ hk, const __bf16* __restrict__ hv,
    const __bf16* __restrict__ tw2b, const float* __restrict__ tb2,
    float* __restrict__ tk, float* __restrict__ tv)
{
  const int wave = threadIdx.x >> 5, lane = threadIdx.x & 31;
  const __bf16* A = blockIdx.y ? hv : hk;
  float* O = blockIdx.y ? tv : tk;
  const int mt = (wave & 1) * 16;
  const int ncol0 = blockIdx.x * 64 + (wave >> 1) * 16;
  v8f acc = gemm32_tile(A, tw2b, H_DIM, mt, ncol0, lane);
  const int col = ncol0 + (lane & 15);
  const float bias = tb2[col];
  const int rb = mt + ((lane >= 16) ? 8 : 0);
#pragma unroll
  for (int i = 0; i < 8; ++i)
    O[(size_t)(rb + i) * H_DIM + col] = acc[i] + bias;
}

// ---------------------------------------------------------------- kernel 5
__global__ __launch_bounds__(256) void output_kernel(
    const float* __restrict__ keys, const float* __restrict__ values,
    const float* __restrict__ imp, const int* __restrict__ idx,
    const float* __restrict__ tk, const float* __restrict__ tv,
    float* __restrict__ outk, float* __restrict__ outv)
{
  const size_t i = ((size_t)blockIdx.x * 256 + threadIdx.x) * 4;
  const size_t row = i >> 11;                 // / H
  const int h = (int)(i & (H_DIM - 1));
  const int c = idx[row];
  const bool pass = imp[row] > 0.1f;          // SPARSITY_THRESHOLD
  const float4 k4 = *(const float4*)(keys + i);
  const float4 v4 = *(const float4*)(values + i);
  const float4 ck = *(const float4*)(tk + (size_t)c * H_DIM + h);
  const float4 cv = *(const float4*)(tv + (size_t)c * H_DIM + h);
  *(float4*)(outk + i) = pass ? k4 : ck;
  *(float4*)(outv + i) = pass ? v4 : cv;
}

// ---------------------------------------------------------------- launcher
extern "C" void kernel_launch(void* const* d_in, const int* in_sizes, int n_in,
                              void* d_out, int out_size, void* d_ws, size_t ws_size,
                              hipStream_t stream)
{
  const float* keys    = (const float*)d_in[0];
  const float* values  = (const float*)d_in[1];
  const float* imp     = (const float*)d_in[2];
  const float* cw1     = (const float*)d_in[3];
  const float* cb1     = (const float*)d_in[4];
  const float* cw2     = (const float*)d_in[5];
  const float* cb2     = (const float*)d_in[6];
  const float* tw1     = (const float*)d_in[7];
  const float* tb1     = (const float*)d_in[8];
  const float* tw2     = (const float*)d_in[9];
  const float* tb2     = (const float*)d_in[10];
  const float* noise_k = (const float*)d_in[11];
  const float* noise_v = (const float*)d_in[12];
  float* outk = (float*)d_out;
  float* outv = outk + (size_t)N_ROWS * H_DIM;

  char* ws = (char*)d_ws;
  size_t o = 0;
  auto alloc = [&](size_t b) { char* p = ws + o; o = (o + b + 255) & ~(size_t)255; return p; };
  __bf16* cw1b   = (__bf16*)alloc((size_t)HH_DIM * H_DIM * 2);
  __bf16* cw2b   = (__bf16*)alloc((size_t)C_DIM * HH_DIM * 2);
  __bf16* tw1b   = (__bf16*)alloc((size_t)H_DIM * H_DIM * 2);
  __bf16* tw2b   = (__bf16*)alloc((size_t)H_DIM * H_DIM * 2);
  int*    idx    = (int*)  alloc((size_t)N_ROWS * 4);
  float*  ksum   = (float*)alloc((size_t)C_DIM * H_DIM * 4);
  float*  vsum   = (float*)alloc((size_t)C_DIM * H_DIM * 4);
  float*  counts = (float*)alloc(256);
  __bf16* kcb    = (__bf16*)alloc((size_t)C_DIM * H_DIM * 2);
  __bf16* vcb    = (__bf16*)alloc((size_t)C_DIM * H_DIM * 2);
  __bf16* hkb    = (__bf16*)alloc((size_t)C_DIM * H_DIM * 2);
  __bf16* hvb    = (__bf16*)alloc((size_t)C_DIM * H_DIM * 2);
  float*  tk     = (float*)alloc((size_t)C_DIM * H_DIM * 4);
  float*  tv     = (float*)alloc((size_t)C_DIM * H_DIM * 4);

  // 1) weights fp32 -> bf16 (L2-resident thereafter)
  cvt_kernel<<<(HH_DIM * H_DIM) / 256, 256, 0, stream>>>(cw1, cw1b, HH_DIM * H_DIM);
  cvt_kernel<<<(C_DIM * HH_DIM) / 256, 256, 0, stream>>>(cw2, cw2b, C_DIM * HH_DIM);
  cvt_kernel<<<(H_DIM * H_DIM) / 256, 256, 0, stream>>>(tw1, tw1b, H_DIM * H_DIM);
  cvt_kernel<<<(H_DIM * H_DIM) / 256, 256, 0, stream>>>(tw2, tw2b, H_DIM * H_DIM);

  // 2) zero accumulators (ksum, vsum, counts are contiguous in ws)
  zero_kernel<<<(2 * C_DIM * H_DIM + 64 + 255) / 256, 256, 0, stream>>>(
      ksum, 2 * C_DIM * H_DIM + 64);

  // 3) fused MLP-projection + argmax assignment
  constexpr size_t assign_lds =
      (size_t)MT * KP * 2 + (size_t)MT * HS * 2 + (size_t)MT * 32 * 4;
  assign_kernel<<<N_ROWS / MT, 256, assign_lds, stream>>>(keys, cw1b, cb1, cw2b,
                                                          cb2, idx);

  // 4) segment reduction (counts, ksum, vsum)
  segsum_kernel<<<128, 256, (size_t)2 * C_DIM * 512 * 4, stream>>>(
      keys, values, idx, ksum, vsum, counts, N_ROWS / 128);

  // 5) centroid means + empty reseed -> bf16 centroids
  finalize_kernel<<<(C_DIM * H_DIM) / 256, 256, 0, stream>>>(
      ksum, vsum, counts, noise_k, noise_v, kcb, vcb);

  // 6) centroid MLP transform (only 32 rows each!)
  transform1_kernel<<<dim3(H_DIM / 64, 2), 256, 0, stream>>>(kcb, vcb, tw1b, tb1,
                                                             hkb, hvb);
  transform2_kernel<<<dim3(H_DIM / 64, 2), 256, 0, stream>>>(hkb, hvb, tw2b, tb2,
                                                             tk, tv);

  // 7) gather + importance passthrough (bandwidth-bound epilogue)
  output_kernel<<<(int)((size_t)N_ROWS * H_DIM / 4 / 256), 256, 0, stream>>>(
      keys, values, imp, idx, tk, tv, outk, outv);

  (void)in_sizes; (void)n_in; (void)out_size; (void)ws_size;
}
